// GraphAttentionEmbedder_5239860101364
// MI455X (gfx1250) — matrix operand order repaired and checked by reading, not compiled, not run on hardware
//
#include <hip/hip_runtime.h>
#include <hip/hip_bf16.h>

// ---------------------------------------------------------------------------
// MI455X (gfx1250) fused temporal-graph-attention embedder.
// bf16 WMMA (v_wmma_f32_16x16x32_bf16) for all GEMMs; associativity trick
// removes the big K/V projections:   scores = kv @ (k_w^T @ Q_h^T)
//                                    ctx    = (attn @ kv) @ v_w^T
// k_b is softmax-invariant (adds a per-(q,h) constant to every key score);
// v_b is added after the ctx projection since attention weights sum to 1.
// ---------------------------------------------------------------------------

typedef __attribute__((ext_vector_type(16))) __bf16 bf16x16;
typedef __attribute__((ext_vector_type(8)))  __bf16 bf16x8;
typedef __attribute__((ext_vector_type(8)))  float  f32x8;

#define QT    8                 // queries per block
#define KN    20                // neighbors per query
#define RKV   (QT * KN)         // 160 kv rows per block
#define DN    128
#define DT    128
#define DE    128
#define EDIM  256               // query embed dim
#define KVD   384               // kv embed dim
#define HD    128               // head dim (2 heads)
#define DOUT  128

// LDS strides (elements). 392*2B = 784B rows -> 4-bank skew, conflict-free.
#define SKV_ST 392
#define WEF_ST 392
#define CKV_ST 392
#define MER_ST 392
#define QP_ST  264
#define H1_ST  136

// ---- WMMA fragment helpers (ISA 7.12.2 layouts, wave32) -------------------
// A (16x32 bf16): lane<16 holds M=lane, K chunks [h*8 .. h*8+7] and [16+h*8..]
static __device__ __forceinline__ bf16x16
ldfragA(const __bf16* __restrict__ base, int lane, int stride) {
    int r = lane & 15, h = lane >> 4;
    const __bf16* p0 = base + (size_t)r * stride + h * 8;
    bf16x8 lo = *(const bf16x8*)p0;
    bf16x8 hi = *(const bf16x8*)(p0 + 16);
    bf16x16 v;
#pragma unroll
    for (int i = 0; i < 8; ++i) { v[i] = lo[i]; v[i + 8] = hi[i]; }
    return v;
}

// B (32x16 bf16) stored as [N][K] row-major: lane holds column n=lane&15,
// K chunks [h*8 .. h*8+7] and [16+h*8 ..], h = lane>>4 (same as A).
static __device__ __forceinline__ bf16x16
ldfragB(const __bf16* __restrict__ base, int lane, int stride) {
    int n = lane & 15, h = lane >> 4;
    const __bf16* p0 = base + (size_t)n * stride + h * 8;
    bf16x8 lo = *(const bf16x8*)p0;
    bf16x8 hi = *(const bf16x8*)(p0 + 16);
    bf16x16 v;
#pragma unroll
    for (int i = 0; i < 8; ++i) { v[i] = lo[i]; v[i + 8] = hi[i]; }
    return v;
}

static __device__ __forceinline__ f32x8
wmma_bf16(bf16x16 a, bf16x16 b, f32x8 c) {
    return __builtin_amdgcn_wmma_f32_16x16x32_bf16(
        false, a, false, b, (short)0, c, false, false);
}

// ---- prep: f32 -> bf16 weight conversion (optionally transposed) ----------
__global__ void tgat_cvt_kernel(const float* __restrict__ src,
                                __bf16* __restrict__ dst,
                                int rows, int cols, int transpose) {
    int i = blockIdx.x * blockDim.x + threadIdx.x;
    int n = rows * cols;
    if (i >= n) return;
    float v = src[i];
    if (transpose) {
        int r = i / cols, c = i - r * cols;
        dst[(size_t)c * rows + r] = (__bf16)v;
    } else {
        dst[i] = (__bf16)v;
    }
}

// ---- main fused kernel ----------------------------------------------------
__global__ __launch_bounds__(256) void tgat_main_kernel(
    const int* __restrict__ idx, const float* __restrict__ t,
    const int* __restrict__ nbrs, const float* __restrict__ e_t,
    const int* __restrict__ e_id, const unsigned char* __restrict__ mask,
    const float* __restrict__ nodes, const float* __restrict__ memv,
    const float* __restrict__ events,
    const float* __restrict__ time_w, const float* __restrict__ time_b,
    const float* __restrict__ q_b, const float* __restrict__ v_b,
    const float* __restrict__ out_b,
    const float* __restrict__ mlp_b1, const float* __restrict__ mlp_b2,
    const __bf16* __restrict__ wQ,   // q_w     bf16 [256][256]
    const __bf16* __restrict__ wKT,  // k_w^T   bf16 [384][256]
    const __bf16* __restrict__ wV,   // v_w     bf16 [256][384]
    const __bf16* __restrict__ wO,   // out_w   bf16 [256][256]
    const __bf16* __restrict__ wM1,  // mlp_w1  bf16 [128][384]
    const __bf16* __restrict__ wM2,  // mlp_w2  bf16 [128][128]
    float* __restrict__ out, int Ntot)
{
    // persistent LDS
    __shared__ __align__(16) __bf16 sKV[RKV * SKV_ST];          // 125,440 B
    __shared__ __align__(16) unsigned char sBufA[32 * WEF_ST * 2]; // 25,088 B
    __shared__ __align__(16) unsigned char sBufB[16 * QP_ST * 2];  //  8,448 B
    __shared__ __align__(16) __bf16 sQfeat[QT * DN];            //   2,048 B
    __shared__ float sAttn[QT * 2 * KN];                        //   1,280 B
    __shared__ int   sNoNb[QT];

    // overlay buffers (phases strictly alternate A/B with barriers between)
    __bf16* sQfull  = (__bf16*)sBufA;   // [16][QP_ST]  P1-P2
    __bf16* sWeffT  = (__bf16*)sBufA;   // [32][WEF_ST] P3-P4
    __bf16* sCtxKV  = (__bf16*)sBufA;   // [2][16][CKV_ST] P5b-P6
    __bf16* sMerged = (__bf16*)sBufA;   // [16][MER_ST] P7-P8
    __bf16* sQp     = (__bf16*)sBufB;   // [16][QP_ST]  P2-P3
    float*  sRawS   = (float*)sBufB;    // [QT][2][KN]  P4-P5a
    __bf16* sCtx    = (__bf16*)sBufB;   // [16][QP_ST]  P6-P7
    __bf16* sH1     = (__bf16*)sBufB;   // [16][H1_ST]  P8-P9

    const int tid  = threadIdx.x;
    const int lane = tid & 31;
    const int wave = tid >> 5;
    const int q0   = blockIdx.x * QT;

    // ---------------- Phase 1: gather + build q_full / kv_full -------------
    if (tid < QT) {
        int qg = q0 + tid; if (qg >= Ntot) qg = Ntot - 1;
        int any = 0;
        const unsigned char* mp = mask + (size_t)qg * KN;
#pragma unroll
        for (int k = 0; k < KN; ++k) any |= mp[k];
        sNoNb[tid] = !any;
    }
    for (int e = tid; e < 16 * EDIM; e += 256) {
        int m = e >> 8, c = e & 255;
        float v = 0.f;
        if (m < QT) {
            int qg = q0 + m; if (qg >= Ntot) qg = Ntot - 1;
            if (c < DN) {
                size_t node = (size_t)idx[qg];
                v = nodes[node * DN + c] + memv[node * DN + c];
                sQfeat[m * DN + c] = (__bf16)v;
            } else {
                v = cosf(time_b[c - DN]);   // query time delta == 0
            }
        }
        sQfull[m * QP_ST + c] = (__bf16)v;
    }
    for (int r = wave; r < RKV; r += 8) {
        int q = r / KN, k = r - q * KN;
        int qg = q0 + q; if (qg >= Ntot) qg = Ntot - 1;
        size_t base = (size_t)qg * KN + k;
        size_t nb = (size_t)nbrs[base];
        size_t ev = (size_t)e_id[base];
        float  mk = mask[base] ? 1.f : 0.f;
        float  dt = t[qg] - e_t[base];
        __bf16* row = sKV + (size_t)r * SKV_ST;
        for (int c = lane; c < KVD; c += 32) {
            float v;
            if (c < DN)           v = mk * (nodes[nb * DN + c] + memv[nb * DN + c]);
            else if (c < DN + DT) { int d = c - DN; v = cosf(dt * time_w[d] + time_b[d]); }
            else                  v = events[ev * DE + (c - DN - DT)];
            row[c] = (__bf16)v;
        }
    }
    __syncthreads();

    // ---------------- Phase 2: Qp = q_full @ q_w^T + q_b  (16x256x256) -----
    for (int nt = wave; nt < 16; nt += 8) {
        f32x8 acc = {};
#pragma unroll
        for (int ks = 0; ks < EDIM / 32; ++ks) {
            bf16x16 a = ldfragA(sQfull + ks * 32, lane, QP_ST);
            bf16x16 b = ldfragB(wQ + (size_t)(nt * 16) * EDIM + ks * 32, lane, EDIM);
            acc = wmma_bf16(a, b, acc);
        }
        int n = nt * 16 + (lane & 15);
        float bias = q_b[n];
#pragma unroll
        for (int v = 0; v < 8; ++v) {
            int m = (lane >> 4) * 8 + v;
            sQp[m * QP_ST + n] = (__bf16)(acc[v] + bias);
        }
    }
    __syncthreads();

    // ---------------- Phase 3: Weff^T[h*16+q][c] = sum_d Qp[q][h,d]*k_w[h,d][c]
    for (int job = wave; job < 48; job += 8) {        // 2 heads * 24 n-tiles
        int h = job / 24, nt = job - h * 24;
        f32x8 acc = {};
#pragma unroll
        for (int ks = 0; ks < HD / 32; ++ks) {
            bf16x16 a = ldfragA(sQp + h * HD + ks * 32, lane, QP_ST);
            bf16x16 b = ldfragB(wKT + (size_t)(nt * 16) * EDIM + h * HD + ks * 32,
                                lane, EDIM);
            acc = wmma_bf16(a, b, acc);
        }
        int c = nt * 16 + (lane & 15);
#pragma unroll
        for (int v = 0; v < 8; ++v) {
            int m = (lane >> 4) * 8 + v;
            sWeffT[(h * 16 + m) * WEF_ST + c] = (__bf16)acc[v];
        }
    }
    __syncthreads();

    // ---------------- Phase 4: scores = kv @ Weff  (160 x 32 x 384) --------
    for (int job = wave; job < 20; job += 8) {        // 10 m-tiles * 2 n-tiles
        int mt = job >> 1, nt = job & 1;
        f32x8 acc = {};
#pragma unroll
        for (int ks = 0; ks < KVD / 32; ++ks) {
            bf16x16 a = ldfragA(sKV + (size_t)(mt * 16) * SKV_ST + ks * 32, lane, SKV_ST);
            bf16x16 b = ldfragB(sWeffT + (size_t)(nt * 16) * WEF_ST + ks * 32, lane, WEF_ST);
            acc = wmma_bf16(a, b, acc);
        }
        int j  = nt * 16 + (lane & 15);               // j = h*16 + q
        int qj = j & 15, hj = j >> 4;
#pragma unroll
        for (int v = 0; v < 8; ++v) {
            int rr = mt * 16 + (lane >> 4) * 8 + v;   // kv row
            int q = rr / KN, k = rr - q * KN;
            if (q == qj)                              // keep block-diagonal only
                sRawS[(q * 2 + hj) * KN + k] = acc[v];
        }
    }
    __syncthreads();

    // ---------------- Phase 5a: masked softmax over 20 neighbors -----------
    if (tid < QT * 2) {
        int q = tid & (QT - 1), h = tid >> 3;
        int qg = q0 + q; if (qg >= Ntot) qg = Ntot - 1;
        int noNb = sNoNb[q];
        const float scale = 0.088388347648318447f;    // 1/sqrt(128)
        float sc[KN], mx = -3.0e38f;
#pragma unroll
        for (int k = 0; k < KN; ++k) {
            int valid = mask[(size_t)qg * KN + k] || noNb;
            float s = valid ? sRawS[(q * 2 + h) * KN + k] * scale : -3.0e38f;
            sc[k] = s; mx = fmaxf(mx, s);
        }
        float sum = 0.f;
#pragma unroll
        for (int k = 0; k < KN; ++k) { sc[k] = expf(sc[k] - mx); sum += sc[k]; }
        float inv = 1.f / sum;
#pragma unroll
        for (int k = 0; k < KN; ++k) sAttn[(q * 2 + h) * KN + k] = sc[k] * inv;
    }
    __syncthreads();

    // ---------------- Phase 5b: ctxkv[h][q][c] = sum_k attn * kv -----------
    for (int e = tid; e < 2 * 16 * KVD; e += 256) {
        int h = e / (16 * KVD);
        int rem = e - h * 16 * KVD;
        int m = rem / KVD, c = rem - m * KVD;
        float v = 0.f;
        if (m < QT) {
            const float*  ap = sAttn + (m * 2 + h) * KN;
            const __bf16* kp = sKV + (size_t)(m * KN) * SKV_ST + c;
#pragma unroll
            for (int k = 0; k < KN; ++k) v += ap[k] * (float)kp[k * SKV_ST];
        }
        sCtxKV[(h * 16 + m) * CKV_ST + c] = (__bf16)v;
    }
    __syncthreads();

    // ---------------- Phase 6: ctx = ctxkv @ v_w^T + v_b  (per head) -------
    for (int job = wave; job < 16; job += 8) {        // 2 heads * 8 n-tiles
        int h = job >> 3, nt = job & 7;
        f32x8 acc = {};
#pragma unroll
        for (int ks = 0; ks < KVD / 32; ++ks) {
            bf16x16 a = ldfragA(sCtxKV + (size_t)(h * 16) * CKV_ST + ks * 32, lane, CKV_ST);
            bf16x16 b = ldfragB(wV + (size_t)(h * HD + nt * 16) * KVD + ks * 32, lane, KVD);
            acc = wmma_bf16(a, b, acc);
        }
        int d = h * HD + nt * 16 + (lane & 15);
        float bias = v_b[d];
#pragma unroll
        for (int v = 0; v < 8; ++v) {
            int m = (lane >> 4) * 8 + v;
            sCtx[m * QP_ST + d] = (__bf16)(acc[v] + bias);
        }
    }
    __syncthreads();

    // ---------------- Phase 7: out = ctx @ out_w^T + out_b; merge ----------
    for (int nt = wave; nt < 16; nt += 8) {
        f32x8 acc = {};
#pragma unroll
        for (int ks = 0; ks < EDIM / 32; ++ks) {
            bf16x16 a = ldfragA(sCtx + ks * 32, lane, QP_ST);
            bf16x16 b = ldfragB(wO + (size_t)(nt * 16) * EDIM + ks * 32, lane, EDIM);
            acc = wmma_bf16(a, b, acc);
        }
        int n = nt * 16 + (lane & 15);
        float bias = out_b[n];
#pragma unroll
        for (int v = 0; v < 8; ++v) {
            int m = (lane >> 4) * 8 + v;
            float val = (m < QT && !sNoNb[m]) ? (acc[v] + bias) : 0.f;
            sMerged[m * MER_ST + n] = (__bf16)val;
        }
    }
    for (int e = tid; e < 16 * DN; e += 256) {        // merged[:,256:384]=q_feat
        int m = e >> 7, c = e & 127;
        float v = (m < QT) ? (float)sQfeat[m * DN + c] : 0.f;
        sMerged[m * MER_ST + EDIM + c] = (__bf16)v;
    }
    __syncthreads();

    // ---------------- Phase 8: h1 = relu(merged @ w1^T + b1) ---------------
    for (int nt = wave; nt < 8; nt += 8) {
        f32x8 acc = {};
#pragma unroll
        for (int ks = 0; ks < KVD / 32; ++ks) {
            bf16x16 a = ldfragA(sMerged + ks * 32, lane, MER_ST);
            bf16x16 b = ldfragB(wM1 + (size_t)(nt * 16) * KVD + ks * 32, lane, KVD);
            acc = wmma_bf16(a, b, acc);
        }
        int n = nt * 16 + (lane & 15);
        float bias = mlp_b1[n];
#pragma unroll
        for (int v = 0; v < 8; ++v) {
            int m = (lane >> 4) * 8 + v;
            sH1[m * H1_ST + n] = (__bf16)fmaxf(acc[v] + bias, 0.f);
        }
    }
    __syncthreads();

    // ---------------- Phase 9: out = h1 @ w2^T + b2  -> global -------------
    for (int nt = wave; nt < 8; nt += 8) {
        f32x8 acc = {};
#pragma unroll
        for (int ks = 0; ks < DOUT / 32; ++ks) {
            bf16x16 a = ldfragA(sH1 + ks * 32, lane, H1_ST);
            bf16x16 b = ldfragB(wM2 + (size_t)(nt * 16) * DOUT + ks * 32, lane, DOUT);
            acc = wmma_bf16(a, b, acc);
        }
        int n = nt * 16 + (lane & 15);
        float bias = mlp_b2[n];
#pragma unroll
        for (int v = 0; v < 8; ++v) {
            int m = (lane >> 4) * 8 + v;
            int row = q0 + m;
            if (m < QT && row < Ntot)
                out[(size_t)row * DOUT + n] = acc[v] + bias;
        }
    }
}

// ---------------------------------------------------------------------------
extern "C" void kernel_launch(void* const* d_in, const int* in_sizes, int n_in,
                              void* d_out, int out_size, void* d_ws, size_t ws_size,
                              hipStream_t stream)
{
    const int*           idx    = (const int*)          d_in[0];
    const float*         t      = (const float*)        d_in[1];
    const int*           nbrs   = (const int*)          d_in[2];
    const float*         e_t    = (const float*)        d_in[3];
    const int*           e_id   = (const int*)          d_in[4];
    const unsigned char* mask   = (const unsigned char*)d_in[5];
    const float*         nodes  = (const float*)        d_in[6];
    const float*         memv   = (const float*)        d_in[7];
    const float*         events = (const float*)        d_in[8];
    const float*         time_w = (const float*)        d_in[9];
    const float*         time_b = (const float*)        d_in[10];
    const float*         q_w    = (const float*)        d_in[11];
    const float*         q_b    = (const float*)        d_in[12];
    const float*         k_w    = (const float*)        d_in[13];
    /* k_b (d_in[14]) intentionally unused: softmax-invariant               */
    const float*         v_w    = (const float*)        d_in[15];
    const float*         v_b    = (const float*)        d_in[16];
    const float*         out_w  = (const float*)        d_in[17];
    const float*         out_b  = (const float*)        d_in[18];
    const float*         w1     = (const float*)        d_in[19];
    const float*         b1     = (const float*)        d_in[20];
    const float*         w2     = (const float*)        d_in[21];
    const float*         b2     = (const float*)        d_in[22];

    const int N = in_sizes[0];

    // bf16 weight cache in workspace (786,432 bytes total)
    __bf16* ws  = (__bf16*)d_ws;
    __bf16* wQ  = ws;            // [256][256]
    __bf16* wKT = ws + 65536;    // [384][256]  (k_w transposed)
    __bf16* wV  = ws + 163840;   // [256][384]
    __bf16* wO  = ws + 262144;   // [256][256]
    __bf16* wM1 = ws + 327680;   // [128][384]
    __bf16* wM2 = ws + 376832;   // [128][128]

    struct CvtJob { const float* s; __bf16* d; int r, c, tr; };
    const CvtJob jobs[6] = {
        { q_w,   wQ,  256, 256, 0 },
        { k_w,   wKT, 256, 384, 1 },
        { v_w,   wV,  256, 384, 0 },
        { out_w, wO,  256, 256, 0 },
        { w1,    wM1, 128, 384, 0 },
        { w2,    wM2, 128, 128, 0 },
    };
    for (int i = 0; i < 6; ++i) {
        int n = jobs[i].r * jobs[i].c;
        tgat_cvt_kernel<<<(n + 255) / 256, 256, 0, stream>>>(
            jobs[i].s, jobs[i].d, jobs[i].r, jobs[i].c, jobs[i].tr);
    }

    int blocks = (N + QT - 1) / QT;
    tgat_main_kernel<<<blocks, 256, 0, stream>>>(
        idx, t, nbrs, e_t, e_id, mask, nodes, memv, events,
        time_w, time_b, q_b, v_b, out_b, b1, b2,
        wQ, wKT, wV, wO, wM1, wM2,
        (float*)d_out, N);
}
